// KANLayer_76819785056469
// MI455X (gfx1250) — matrix operand order repaired and checked by reading, not compiled, not run on hardware
//
#include <hip/hip_runtime.h>
#include <hip/hip_bf16.h>

typedef _Float16 v2h  __attribute__((ext_vector_type(2)));
typedef _Float16 v4h  __attribute__((ext_vector_type(4)));
typedef _Float16 v8h  __attribute__((ext_vector_type(8)));
typedef _Float16 v16h __attribute__((ext_vector_type(16)));
typedef float    v8f  __attribute__((ext_vector_type(8)));

#define N_ROWS 8192
#define IN_DIM 1024
#define OUT_DIM 1024
#define NB 16
#define KSPLINE (NB * IN_DIM)      // 16384
#define KTOT (KSPLINE + IN_DIM)    // 17408 = 136 * 128
#define BM 128
#define BN 256
#define KB 64
#define PITCH 72                   // halfs per LDS row (pad to spread banks)

// ---------------------------------------------------------------------------
// Phase 1: pack [W | scale_base] (f32) into B^T (f16), row-major [OUT][KTOT].
// ---------------------------------------------------------------------------
__global__ __launch_bounds__(256) void pack_b_kernel(
    const float* __restrict__ W, const float* __restrict__ sb,
    _Float16* __restrict__ Bws) {
  int idx = blockIdx.x * 256 + threadIdx.x;     // quad index, KTOT/4 per row
  int row = idx / (KTOT / 4);
  int k   = (idx - row * (KTOT / 4)) * 4;
  const float* src = (k < KSPLINE) ? (W + (size_t)row * KSPLINE + k)
                                   : (sb + (size_t)row * IN_DIM + (k - KSPLINE));
  float4 v = *(const float4*)src;
  v4h h = { (_Float16)v.x, (_Float16)v.y, (_Float16)v.z, (_Float16)v.w };
  *(v4h*)(Bws + (size_t)row * KTOT + k) = h;
}

// --------------------------- CDNA5 async helpers ---------------------------
__device__ __forceinline__ unsigned lds_off(const void* p) {
  // addrspace(3) ptrtoint == byte offset from LDS base (what VDST expects)
  return (unsigned)(unsigned long long)(__attribute__((address_space(3))) const void*)p;
}

__device__ __forceinline__ void async_load_b128_to_lds(const void* gptr,
                                                       unsigned loff) {
  asm volatile("global_load_async_to_lds_b128 %0, %1, off"
               :: "v"(loff), "v"(gptr) : "memory");
}

__device__ __forceinline__ void wait_asynccnt0() {
  asm volatile("s_wait_asynccnt 0x0" ::: "memory");
}

// ------------------------------ stage pieces -------------------------------
__device__ __forceinline__ void stage_b_async(const _Float16* __restrict__ Bws,
                                              int n0, int k0, int t,
                                              _Float16* Bs) {
#pragma unroll
  for (int it = 0; it < 4; ++it) {
    int c   = t + it * 256;         // 16B chunk id, 4 chunks per row
    int row = c >> 2;
    int ch  = (c & 3) * 8;
    const void* g = Bws + (size_t)(n0 + row) * KTOT + k0 + ch;
    async_load_b128_to_lds(g, lds_off(&Bs[row * PITCH + ch]));
  }
}

__device__ __forceinline__ void compute_a(const float* __restrict__ x,
                                          const float* __restrict__ bw,
                                          const float* __restrict__ bb,
                                          int m0, int k0, int t,
                                          _Float16* As) {
  const float C2   = -23.083120654223414f;   // -BETA * log2(e)
  const float R2PI = 0.15915494309189535f;   // 1/(2*pi) for v_cos_f32
  if (k0 < KSPLINE) {               // uniform scalar branch per stage
    const int j  = k0 >> 10;
    const int i0 = k0 & (IN_DIM - 1);
#pragma unroll
    for (int e = 0; e < 16; ++e) {
      int p   = t + e * 256;        // pair id 0..4095
      int row = p >> 5;             // 32 pairs per 64-wide row
      int kp  = (p & 31) << 1;
      int i   = i0 + kp;
      float2 xv = *(const float2*)(x + (size_t)(m0 + row) * IN_DIM + i);
      float2 wv = *(const float2*)(bw + (size_t)j * IN_DIM + i);
      float2 bv = *(const float2*)(bb + (size_t)j * IN_DIM + i);
      float t0 = fmaf(xv.x, wv.x, bv.x);
      float t1 = fmaf(xv.y, wv.y, bv.y);
      float a0 = __builtin_amdgcn_exp2f(C2 * t0 * t0);  // v_exp_f32 (TRANS)
      float a1 = __builtin_amdgcn_exp2f(C2 * t1 * t1);
      *(v2h*)&As[row * PITCH + kp] = (v2h){ (_Float16)a0, (_Float16)a1 };
    }
  } else {
    const int i0 = k0 - KSPLINE;
#pragma unroll
    for (int e = 0; e < 16; ++e) {
      int p   = t + e * 256;
      int row = p >> 5;
      int kp  = (p & 31) << 1;
      int i   = i0 + kp;
      float2 xv = *(const float2*)(x + (size_t)(m0 + row) * IN_DIM + i);
      float a0 = __builtin_amdgcn_cosf(xv.x * R2PI);    // v_cos_f32 (TRANS)
      float a1 = __builtin_amdgcn_cosf(xv.y * R2PI);
      *(v2h*)&As[row * PITCH + kp] = (v2h){ (_Float16)a0, (_Float16)a1 };
    }
  }
}

// fragment loaders (layouts per CDNA5 ISA §7.12.2)
__device__ __forceinline__ v16h load_afrag(const _Float16* As, int row,
                                           int koff) {
  // A 16x32 f16: lane<16 holds K 0..7 & 16..23; lane>=16 holds +8
  v8h lo = *(const v8h*)&As[row * PITCH + koff];        // ds_load_b128
  v8h hi = *(const v8h*)&As[row * PITCH + koff + 16];   // ds_load_b128
  return __builtin_shufflevector(lo, hi, 0,1,2,3,4,5,6,7,
                                         8,9,10,11,12,13,14,15);
}

__device__ __forceinline__ v16h load_bfrag(const _Float16* Bs, int row,
                                           int koff) {
  // B 32x16 f16: lane<16 holds K 0..7 & 16..23; lane>=16 holds +8
  v8h lo = *(const v8h*)&Bs[row * PITCH + koff];
  v8h hi = *(const v8h*)&Bs[row * PITCH + koff + 16];
  return __builtin_shufflevector(lo, hi, 0,1,2,3,4,5,6,7,
                                         8,9,10,11,12,13,14,15);
}

__device__ __forceinline__ void mma_stage(const _Float16* As,
                                          const _Float16* Bs,
                                          int wm, int wn, int l15, int lhi,
                                          v8f (&acc)[2][8]) {
  const int arow0 = wm * 32 + l15;
  const int arow1 = arow0 + 16;
  const int brow0 = wn * 128 + l15;
#pragma unroll
  for (int ks = 0; ks < 2; ++ks) {
    const int kkA = ks * 32 + lhi * 8;
    const int kkB = ks * 32 + lhi * 8;
    v16h a0 = load_afrag(As, arow0, kkA);
    v16h a1 = load_afrag(As, arow1, kkA);
    // software-pipelined B fragments: load nt+1 before consuming nt
    v16h bc = load_bfrag(Bs, brow0, kkB);
#pragma unroll
    for (int nt = 0; nt < 8; ++nt) {
      v16h bn;
      if (nt < 7) bn = load_bfrag(Bs, brow0 + (nt + 1) * 16, kkB);
      acc[0][nt] = __builtin_amdgcn_wmma_f32_16x16x32_f16(
          false, a0, false, bc, (short)0, acc[0][nt], false, false);
      acc[1][nt] = __builtin_amdgcn_wmma_f32_16x16x32_f16(
          false, a1, false, bc, (short)0, acc[1][nt], false, false);
      if (nt < 7) bc = bn;
    }
  }
}

// ---------------------------------------------------------------------------
// Phase 2: fused activation + WMMA GEMM, double-buffered + async-to-LDS.
// ---------------------------------------------------------------------------
__global__ __launch_bounds__(256) void kan_wmma_kernel(
    const float* __restrict__ x, const float* __restrict__ bw,
    const float* __restrict__ bb, const _Float16* __restrict__ Bws,
    const float* __restrict__ bias, float* __restrict__ out) {
  __shared__ _Float16 As[2][BM * PITCH];   // 2 x 18 KB
  __shared__ _Float16 Bs[2][BN * PITCH];   // 2 x 36 KB

  const int t    = threadIdx.x;
  const int lane = t & 31;
  const int wid  = t >> 5;
  const int wm   = wid & 3;   // 0..3 -> 32-row strip
  const int wn   = wid >> 2;  // 0..1 -> 128-col strip
  const int l15  = lane & 15;
  const int lhi  = lane >> 4; // 0/1

  const int m0 = blockIdx.y * BM;
  const int n0 = blockIdx.x * BN;

  v8f acc[2][8] = {};

  // ---- prologue: fill buffer 0 ----
  stage_b_async(Bws, n0, 0, t, Bs[0]);
  compute_a(x, bw, bb, m0, 0, t, As[0]);
  wait_asynccnt0();
  __syncthreads();

  // ---- pipelined main loop, 2 stages per trip (KTOT % 128 == 0) ----
  for (int k0 = 0; k0 < KTOT; k0 += 2 * KB) {
    // even stage: consume buf0, produce buf1 for k0+KB
    stage_b_async(Bws, n0, k0 + KB, t, Bs[1]);
    compute_a(x, bw, bb, m0, k0 + KB, t, As[1]);
    mma_stage(As[0], Bs[0], wm, wn, l15, lhi, acc);
    wait_asynccnt0();
    __syncthreads();

    // odd stage: consume buf1, produce buf0 for k0+2*KB (if any)
    if (k0 + 2 * KB < KTOT) {
      stage_b_async(Bws, n0, k0 + 2 * KB, t, Bs[0]);
      compute_a(x, bw, bb, m0, k0 + 2 * KB, t, As[0]);
    }
    mma_stage(As[1], Bs[1], wm, wn, l15, lhi, acc);
    wait_asynccnt0();
    __syncthreads();
  }

  // ---- epilogue: bias add + store (C/D layout: VGPR v -> M=v/v+8) ----
#pragma unroll
  for (int nt = 0; nt < 8; ++nt) {
    int col  = n0 + wn * 128 + nt * 16 + l15;
    float bv = bias[col];
#pragma unroll
    for (int mt = 0; mt < 2; ++mt) {
      int mbase = m0 + wm * 32 + mt * 16 + lhi * 8;
#pragma unroll
      for (int v = 0; v < 8; ++v) {
        out[(size_t)(mbase + v) * OUT_DIM + col] = acc[mt][nt][v] + bv;
      }
    }
  }
}

extern "C" void kernel_launch(void* const* d_in, const int* in_sizes, int n_in,
                              void* d_out, int out_size, void* d_ws, size_t ws_size,
                              hipStream_t stream) {
  const float* x    = (const float*)d_in[0];
  const float* bw   = (const float*)d_in[1];
  const float* bb   = (const float*)d_in[2];
  const float* W    = (const float*)d_in[3];
  const float* bias = (const float*)d_in[4];
  const float* sb   = (const float*)d_in[5];
  float* out = (float*)d_out;

  _Float16* Bws = (_Float16*)d_ws;   // needs OUT_DIM*KTOT*2 = 35.7 MB scratch

  pack_b_kernel<<<(OUT_DIM * KTOT / 4) / 256, 256, 0, stream>>>(W, sb, Bws);

  dim3 grid(OUT_DIM / BN, N_ROWS / BM);   // (4, 64)
  kan_wmma_kernel<<<grid, 256, 0, stream>>>(x, bw, bb, Bws, bias, out);
}
